// FastASTEncoder_49512382988924
// MI455X (gfx1250) — matrix operand order repaired and checked by reading, not compiled, not run on hardware
//
#include <hip/hip_runtime.h>
#include <hip/hip_bf16.h>
#include <math.h>

// ---------------- problem constants (match reference) ----------------
#define B_    16
#define N_    512
#define DM_   512
#define H_    8
#define PARH_ 4
#define DK_   64
#define R_    8
#define L_    2
#define DFF_  2048
#define EPS_  1e-5f
#define MT_   (B_ * N_)        // 8192 tokens
#define SCALE_ 0.125f          // 1/sqrt(64)

typedef _Float16 half_t;
typedef __attribute__((ext_vector_type(16))) _Float16 v16h;
typedef __attribute__((ext_vector_type(8)))  _Float16 v8h;
typedef __attribute__((ext_vector_type(8)))  float    v8f;

// CDNA5 async global->LDS path (guarded; falls back to sync copies)
#if __has_builtin(__builtin_amdgcn_global_load_async_to_lds_b128) && \
    __has_builtin(__builtin_amdgcn_s_wait_asynccnt)
#define HAS_ASYNC 1
typedef int v4i __attribute__((vector_size(16)));
typedef __attribute__((address_space(1))) v4i gv4i_t;   // global int4
typedef __attribute__((address_space(3))) v4i lv4i_t;   // LDS int4
#else
#define HAS_ASYNC 0
#endif

// ---------------- WMMA GEMM ----------------
// A: f16 [M,K] row-major (K contiguous). Bt: f16 [N,K] row-major (K contiguous).
// C = act(A @ B + bias), written as fp32 (Cf) and/or f16 (Ch).
// Block tile 128x128, BK=32. 8 waves: 4 row-groups x 2 col-groups; 32x64 per wave.
#define BM 128
#define BN 128
#define BK 32
#define LDP (BK + 8)   // LDS row stride in halves: 80B rows, 16B-aligned, conflict-free

__global__ __launch_bounds__(256)
void gemm_wmma_f16(const half_t* __restrict__ A, const half_t* __restrict__ Bt,
                   const float* __restrict__ bias, float* __restrict__ Cf,
                   half_t* __restrict__ Ch, int M, int N, int K, int relu)
{
    __shared__ _Float16 As[2][BM][LDP];

    const int tid  = threadIdx.x;
    const int lane = tid & 31;
    const int wave = tid >> 5;
    const int wr   = wave & 3;      // 0..3 -> 32-row group
    const int wc   = wave >> 2;     // 0..1 -> 64-col group
    const int m0   = blockIdx.y * BM;
    const int n0   = blockIdx.x * BN;
    const int kst  = K >> 5;

    // A-tile staging: 128 rows x 64B = 512 x 16B chunks, 2 per thread
    const int ca = tid * 2;
    const int rowa = ca >> 2, offa = (ca & 3) * 8;          // halves
    const int cb = tid * 2 + 1;
    const int rowb = cb >> 2, offb = (cb & 3) * 8;

    auto stage = [&](int buf, int k0) {
#if HAS_ASYNC
        __builtin_amdgcn_global_load_async_to_lds_b128(
            (gv4i_t*)const_cast<half_t*>(A + (size_t)(m0 + rowa) * K + k0 + offa),
            (lv4i_t*)&As[buf][rowa][offa], 0, 0);
        __builtin_amdgcn_global_load_async_to_lds_b128(
            (gv4i_t*)const_cast<half_t*>(A + (size_t)(m0 + rowb) * K + k0 + offb),
            (lv4i_t*)&As[buf][rowb][offb], 0, 0);
#else
        uint4 da = *reinterpret_cast<const uint4*>(A + (size_t)(m0 + rowa) * K + k0 + offa);
        uint4 db = *reinterpret_cast<const uint4*>(A + (size_t)(m0 + rowb) * K + k0 + offb);
        *reinterpret_cast<uint4*>(&As[buf][rowa][offa]) = da;
        *reinterpret_cast<uint4*>(&As[buf][rowb][offb]) = db;
#endif
    };

    v8f acc[2][4] = {};

    stage(0, 0);
    for (int t = 0; t < kst; ++t) {
        const int k0 = t << 5;
        if (t + 1 < kst) stage((t + 1) & 1, (t + 1) << 5);
#if HAS_ASYNC
        if (t + 1 < kst) __builtin_amdgcn_s_wait_asynccnt(2);
        else             __builtin_amdgcn_s_wait_asynccnt(0);
#endif
        __syncthreads();   // tile t visible to all waves

        const _Float16 (*Ab)[LDP] = As[t & 1];
        const int mrow = lane & 15;
        const int hi   = lane >> 4;
        const int kbA  = hi << 4;       // A frag: K 0..15 / 16..31
        v16h af[2];
        #pragma unroll
        for (int f = 0; f < 2; ++f) {
            const _Float16* ap = Ab[wr * 32 + f * 16 + mrow];
            const v8h lo = *reinterpret_cast<const v8h*>(ap + kbA);
            const v8h hh = *reinterpret_cast<const v8h*>(ap + 8 + kbA);
            #pragma unroll
            for (int i = 0; i < 8; ++i) { af[f][i] = lo[i]; af[f][8 + i] = hh[i]; }
        }
        const int kbB = k0 + (hi << 4); // B frag: lanes<16 -> K 0..15 ; >=16 -> 16..31
        #pragma unroll
        for (int j = 0; j < 4; ++j) {
            const int n = n0 + wc * 64 + j * 16 + mrow;
            const v16h bf = *reinterpret_cast<const v16h*>(&Bt[(size_t)n * K + kbB]);
            acc[0][j] = __builtin_amdgcn_wmma_f32_16x16x32_f16(
                false, af[0], false, bf, (short)0, acc[0][j], false, false);
            acc[1][j] = __builtin_amdgcn_wmma_f32_16x16x32_f16(
                false, af[1], false, bf, (short)0, acc[1][j], false, false);
        }
        __syncthreads();   // everyone done reading buf before it is re-filled
    }

    // ---- epilogue: D layout = VGPR r : lanes<16 -> M=r ; lanes>=16 -> M=8+r ; N=lane&15 ----
    const int coln  = lane & 15;
    const int rhalf = (lane >> 4) << 3;
    #pragma unroll
    for (int i = 0; i < 2; ++i) {
        #pragma unroll
        for (int j = 0; j < 4; ++j) {
            const int row_base = m0 + wr * 32 + i * 16 + rhalf;
            const int col      = n0 + wc * 64 + j * 16 + coln;
            const float bv = bias ? bias[col] : 0.0f;
            #pragma unroll
            for (int r = 0; r < 8; ++r) {
                float val = acc[i][j][r] + bv;
                if (relu) val = fmaxf(val, 0.0f);
                const size_t o = (size_t)(row_base + r) * N + col;
                if (Cf) Cf[o] = val;
                if (Ch) Ch[o] = (half_t)val;
            }
        }
    }
}

// ---------------- sparse relational attention: one wave per (b,h,n), f16 in/out ----------------
__global__ __launch_bounds__(256)
void attn_sparse(const half_t* __restrict__ q, const half_t* __restrict__ k,
                 const half_t* __restrict__ v,
                 const int* __restrict__ par, const int* __restrict__ bro,
                 const float* __restrict__ rel_q, const float* __restrict__ rel_k,
                 const float* __restrict__ rel_v, half_t* __restrict__ ctx)
{
    const int lane = threadIdx.x & 31;
    const int w    = blockIdx.x * 8 + (threadIdx.x >> 5);   // 0 .. B*H*N-1
    const int n = w % N_;
    const int h = (w / N_) % H_;
    const int b = w / (N_ * H_);

    const int* edges = (h < PARH_) ? par : bro;
    const size_t qoff = ((size_t)(b * N_ + n)) * DM_ + h * DK_;
    const float q0 = (float)q[qoff + lane];
    const float q1 = (float)q[qoff + 32 + lane];

    float sc[R_], vs0[R_], vs1[R_];
    #pragma unroll
    for (int r = 0; r < R_; ++r) {
        const int m = edges[(b * R_ + r) * N_ + n];
        const size_t ko = ((size_t)(b * N_ + m)) * DM_ + h * DK_;
        const float ks0 = (float)k[ko + lane], ks1 = (float)k[ko + 32 + lane];
        vs0[r] = (float)v[ko + lane];
        vs1[r] = (float)v[ko + 32 + lane];
        const size_t ro = (size_t)(h * R_ + r) * DK_;
        const float rq0 = rel_q[ro + lane], rq1 = rel_q[ro + 32 + lane];
        const float rk0 = rel_k[ro + lane], rk1 = rel_k[ro + 32 + lane];
        // (q + rel_q)·k_s + q·rel_k
        float p = (q0 + rq0) * ks0 + q0 * rk0 + (q1 + rq1) * ks1 + q1 * rk1;
        #pragma unroll
        for (int off = 16; off > 0; off >>= 1) p += __shfl_xor(p, off, 32);
        sc[r] = p * SCALE_;
    }

    float mx = sc[0];
    #pragma unroll
    for (int r = 1; r < R_; ++r) mx = fmaxf(mx, sc[r]);
    float se = 0.0f;
    #pragma unroll
    for (int r = 0; r < R_; ++r) { sc[r] = __expf(sc[r] - mx); se += sc[r]; }
    const float inv = 1.0f / se;

    float c0 = 0.0f, c1 = 0.0f;
    #pragma unroll
    for (int r = 0; r < R_; ++r) {
        const size_t ro = (size_t)(h * R_ + r) * DK_;
        const float a = sc[r] * inv;
        c0 += a * (vs0[r] + rel_v[ro + lane]);
        c1 += a * (vs1[r] + rel_v[ro + 32 + lane]);
    }
    ctx[qoff + lane]      = (half_t)c0;
    ctx[qoff + 32 + lane] = (half_t)c1;
}

// ---------------- fused residual-add + LayerNorm: one wave per token ----------------
// out fp32 (residual stream); out16 optional f16 copy for the next GEMM.
__global__ __launch_bounds__(256)
void add_ln(const float* __restrict__ xin, const float* __restrict__ yin,
            const float* __restrict__ g, const float* __restrict__ bta,
            float* __restrict__ out, half_t* __restrict__ out16)
{
    const int lane = threadIdx.x & 31;
    const int tok  = blockIdx.x * 8 + (threadIdx.x >> 5);
    const size_t base = (size_t)tok * DM_;

    float vals[DM_ / 32];
    float s = 0.0f;
    #pragma unroll
    for (int i = 0; i < DM_ / 32; ++i) {
        const int c = lane + i * 32;
        float t = xin[base + c];
        if (yin) t += yin[base + c];
        vals[i] = t;
        s += t;
    }
    #pragma unroll
    for (int off = 16; off > 0; off >>= 1) s += __shfl_xor(s, off, 32);
    const float mean = s * (1.0f / DM_);

    float var = 0.0f;
    #pragma unroll
    for (int i = 0; i < DM_ / 32; ++i) { const float d = vals[i] - mean; var += d * d; }
    #pragma unroll
    for (int off = 16; off > 0; off >>= 1) var += __shfl_xor(var, off, 32);
    const float rinv = rsqrtf(var * (1.0f / DM_) + EPS_);

    #pragma unroll
    for (int i = 0; i < DM_ / 32; ++i) {
        const int c = lane + i * 32;
        const float o = (vals[i] - mean) * rinv * g[c] + bta[c];
        out[base + c] = o;
        if (out16) out16[base + c] = (half_t)o;
    }
}

// ---------------- one-shot converters ----------------
__global__ __launch_bounds__(256)
void cast_f16(const float* __restrict__ s, half_t* __restrict__ d, int n)
{
    const int i = blockIdx.x * 256 + threadIdx.x;
    if (i < n) d[i] = (half_t)s[i];
}

// src fp32 [K,N] -> dst f16 [N,K] (K contiguous per output column)
__global__ __launch_bounds__(256)
void tpose_cast(const float* __restrict__ s, half_t* __restrict__ d, int K, int N)
{
    const int i = blockIdx.x * 256 + threadIdx.x;
    if (i < K * N) {
        const int kk = i / N, nn = i % N;
        d[(size_t)nn * K + kk] = (half_t)s[i];
    }
}

// ---------------- host launcher ----------------
extern "C" void kernel_launch(void* const* d_in, const int* in_sizes, int n_in,
                              void* d_out, int out_size, void* d_ws, size_t ws_size,
                              hipStream_t stream)
{
    const float* src_emb = (const float*)d_in[0];
    const int*   par     = (const int*)  d_in[1];
    const int*   bro     = (const int*)  d_in[2];
    const float* rel_q   = (const float*)d_in[3];
    const float* rel_k   = (const float*)d_in[4];
    const float* rel_v   = (const float*)d_in[5];
    const float* Wq      = (const float*)d_in[6];
    const float* Wk      = (const float*)d_in[7];
    const float* Wv      = (const float*)d_in[8];
    const float* Wo      = (const float*)d_in[9];
    const float* ln1_g   = (const float*)d_in[10];
    const float* ln1_b   = (const float*)d_in[11];
    const float* W1      = (const float*)d_in[12];
    const float* b1      = (const float*)d_in[13];
    const float* W2      = (const float*)d_in[14];
    const float* b2      = (const float*)d_in[15];
    const float* ln2_g   = (const float*)d_in[16];
    const float* ln2_b   = (const float*)d_in[17];
    const float* normf_g = (const float*)d_in[18];
    const float* normf_b = (const float*)d_in[19];
    float* out = (float*)d_out;

    const size_t TD = (size_t)MT_ * DM_;
    char* w = (char*)d_ws;
    float*  x   = (float*)w;  w += TD * sizeof(float);
    float*  t0  = (float*)w;  w += TD * sizeof(float);
    half_t* x16 = (half_t*)w; w += TD * sizeof(half_t);
    half_t* q16 = (half_t*)w; w += TD * sizeof(half_t);
    half_t* k16 = (half_t*)w; w += TD * sizeof(half_t);
    half_t* v16 = (half_t*)w; w += TD * sizeof(half_t);
    half_t* cx16= (half_t*)w; w += TD * sizeof(half_t);
    half_t* ff16= q16;        // FFN activation (MT_*DFF_ = 4*TD halves) overlays q/k/v/ctx
    half_t* wqT = (half_t*)w; w += (size_t)L_ * DM_ * DM_ * sizeof(half_t);
    half_t* wkT = (half_t*)w; w += (size_t)L_ * DM_ * DM_ * sizeof(half_t);
    half_t* wvT = (half_t*)w; w += (size_t)L_ * DM_ * DM_ * sizeof(half_t);
    half_t* woT = (half_t*)w; w += (size_t)L_ * DM_ * DM_ * sizeof(half_t);
    half_t* w1T = (half_t*)w; w += (size_t)L_ * DM_ * DFF_ * sizeof(half_t);
    half_t* w2T = (half_t*)w; w += (size_t)L_ * DFF_ * DM_ * sizeof(half_t);

    const dim3 blk(256);
    const dim3 gDM(DM_ / BN, MT_ / BM);    // (4, 64)
    const dim3 gFF(DFF_ / BN, MT_ / BM);   // (16, 64)
    const dim3 gAT((B_ * H_ * N_) / 8);
    const dim3 gLN(MT_ / 8);
    const dim3 gCast((int)((TD + 255) / 256));
    const dim3 gTpDM((DM_ * DM_ + 255) / 256);
    const dim3 gTpFF((DM_ * DFF_ + 255) / 256);

    // init residual stream (fp32) + f16 copy; transpose+cast all weights once
    (void)hipMemcpyAsync(x, src_emb, TD * sizeof(float), hipMemcpyDeviceToDevice, stream);
    cast_f16<<<gCast, blk, 0, stream>>>(src_emb, x16, (int)TD);
    for (int l = 0; l < L_; ++l) {
        const size_t sq = (size_t)l * DM_ * DM_;
        const size_t sf = (size_t)l * DM_ * DFF_;
        tpose_cast<<<gTpDM, blk, 0, stream>>>(Wq + sq, wqT + sq, DM_, DM_);
        tpose_cast<<<gTpDM, blk, 0, stream>>>(Wk + sq, wkT + sq, DM_, DM_);
        tpose_cast<<<gTpDM, blk, 0, stream>>>(Wv + sq, wvT + sq, DM_, DM_);
        tpose_cast<<<gTpDM, blk, 0, stream>>>(Wo + sq, woT + sq, DM_, DM_);
        tpose_cast<<<gTpFF, blk, 0, stream>>>(W1 + sf, w1T + sf, DM_, DFF_);
        tpose_cast<<<gTpFF, blk, 0, stream>>>(W2 + sf, w2T + sf, DFF_, DM_);
    }

    for (int l = 0; l < L_; ++l) {
        const size_t sq = (size_t)l * DM_ * DM_;
        const size_t sf = (size_t)l * DM_ * DFF_;

        gemm_wmma_f16<<<gDM, blk, 0, stream>>>(x16, wqT + sq, nullptr, nullptr, q16, MT_, DM_, DM_, 0);
        gemm_wmma_f16<<<gDM, blk, 0, stream>>>(x16, wkT + sq, nullptr, nullptr, k16, MT_, DM_, DM_, 0);
        gemm_wmma_f16<<<gDM, blk, 0, stream>>>(x16, wvT + sq, nullptr, nullptr, v16, MT_, DM_, DM_, 0);

        attn_sparse<<<gAT, blk, 0, stream>>>(q16, k16, v16, par, bro, rel_q, rel_k, rel_v, cx16);

        gemm_wmma_f16<<<gDM, blk, 0, stream>>>(cx16, woT + sq, nullptr, t0, nullptr, MT_, DM_, DM_, 0);
        add_ln<<<gLN, blk, 0, stream>>>(x, t0, ln1_g + l * DM_, ln1_b + l * DM_, x, x16);

        gemm_wmma_f16<<<gFF, blk, 0, stream>>>(x16, w1T + sf, b1 + l * DFF_, nullptr, ff16, MT_, DFF_, DM_, 1);
        gemm_wmma_f16<<<gDM, blk, 0, stream>>>(ff16, w2T + sf, b2 + l * DM_, t0, nullptr, MT_, DM_, DFF_, 0);
        add_ln<<<gLN, blk, 0, stream>>>(x, t0, ln2_g + l * DM_, ln2_b + l * DM_, x, x16);
    }

    add_ln<<<gLN, blk, 0, stream>>>(x, nullptr, normf_g, normf_b, out, nullptr);
}
